// CrossAttentionModel_20684562497797
// MI455X (gfx1250) — matrix operand order repaired and checked by reading, not compiled, not run on hardware
//
#include <hip/hip_runtime.h>
#include <hip/hip_bf16.h>

// ---------------------------------------------------------------------------
// CrossAttention pipeline for MI455X (gfx1250, wave32, WMMA, async-to-LDS).
//   q = v1 @ wq^T ; k = v2 @ wk^T          (bf16 WMMA, fp32 accum)
//   q,k = RMSNorm_fp32(q,k)                (row reduce, wave32 shuffles)
//   s[i,j,h] = <qh[i], kh[j]> / sqrt(128)  (bf16 WMMA over all heads)
//   out[i,j] = mean_h softmax_{axis=1}(s) = mean_h expf(s - s)  (faithful:
//              1 for finite s, NaN-propagating; keeps GEMMs live w/o fastmath)
// Tile staging uses GLOBAL_LOAD_ASYNC_TO_LDS_B128 (ASYNCcnt) so the double
// buffer needs only ONE barrier per K-step and no VGPR staging.
// ---------------------------------------------------------------------------

typedef __attribute__((ext_vector_type(4)))  __bf16 v4bf;
typedef __attribute__((ext_vector_type(8)))  __bf16 v8bf;
typedef __attribute__((ext_vector_type(16))) __bf16 v16bf;
typedef __attribute__((ext_vector_type(8)))  float  v8f;

#define HID   4096
#define NB1   2048
#define NB2   2048
#define HEADS 32

#define TM   128          // workgroup M tile
#define TN   128          // workgroup N tile
#define TK   32           // K per WMMA stage
#define RSB  80           // LDS row stride bytes (64B data + 16B pad, conflict free)
#define STG  (TM * RSB)   // one matrix stage: 10240 B
#define BUFB (2 * STG)    // A+B stage: 20480 B (x2 buffers = 40 KB)
#define KBYTES (TK * 2)   // 64 B advanced per K stage
#define ROWSKIP (64ull * HID * 2ull)  // +64 rows in bytes = 524288

__device__ __forceinline__ void async_b128(unsigned lds, unsigned long long ga) {
  // gfx1250: per-lane 16B global -> LDS, tracked by ASYNCcnt.
  asm volatile("global_load_async_to_lds_b128 %0, %1, off"
               :: "v"(lds), "v"(ga)
               : "memory");
}
__device__ __forceinline__ void wait_async0() {
  asm volatile("s_wait_asynccnt 0x0" ::: "memory");
}

__device__ __forceinline__ v16bf frag_from(const unsigned char* p0,
                                           const unsigned char* p1) {
  v8bf lo = *(const v8bf*)p0;   // ds_load_b128
  v8bf hi = *(const v8bf*)p1;   // ds_load_b128
  return __builtin_shufflevector(lo, hi, 0, 1, 2, 3, 4, 5, 6, 7,
                                         8, 9, 10, 11, 12, 13, 14, 15);
}

// ---------------------------------------------------------------------------
// fp32 -> bf16 bulk convert (b128 load / b64 store)
// ---------------------------------------------------------------------------
__global__ __launch_bounds__(256) void xattn_f32_to_bf16(
    const float* __restrict__ in, __bf16* __restrict__ out, int n4) {
  int i = blockIdx.x * 256 + threadIdx.x;
  if (i >= n4) return;
  float4 v = ((const float4*)in)[i];
  v4bf o;
  o[0] = (__bf16)v.x; o[1] = (__bf16)v.y;
  o[2] = (__bf16)v.z; o[3] = (__bf16)v.w;
  ((v4bf*)out)[i] = o;
}

// ---------------------------------------------------------------------------
// C[m,n] = sum_d A[m,d] * W[n,d]   (x @ W^T), bf16 in, fp32 out.
// 128x128 WG tile, 8 waves (4x2) of 32x64, async-to-LDS double buffer.
// ---------------------------------------------------------------------------
__global__ __launch_bounds__(256) void xattn_gemm_proj(
    const __bf16* __restrict__ A, const __bf16* __restrict__ W,
    float* __restrict__ C) {
  __shared__ unsigned char smem[2 * BUFB];
  const int t    = threadIdx.x;
  const int lane = t & 31;
  const int wid  = t >> 5;
  const int wm   = wid & 3;        // 4 waves along M (32 rows each)
  const int wn   = wid >> 2;       // 2 waves along N (64 cols each)
  const int m0   = blockIdx.y * TM;
  const int n0   = blockIdx.x * TN;
  const int r0   = t >> 2;         // staging row 0..63 (and +64)
  const int cb   = (t & 3) * 16;   // staging byte column
  const int ce   = (t & 3) * 8;    // staging element column
  const int ha   = lane >> 4;      // lane half (0/1)

  // per-lane async staging addresses
  const unsigned sb  = (unsigned)(uintptr_t)smem;   // LDS base offset
  const unsigned la0 = sb + r0 * RSB + cb;
  const unsigned la1 = la0 + 64 * RSB;
  const unsigned lb0 = la0 + STG;
  const unsigned lb1 = la1 + STG;
  const unsigned long long gaA =
      (unsigned long long)(uintptr_t)(A + (size_t)(m0 + r0) * HID + ce);
  const unsigned long long gaB =
      (unsigned long long)(uintptr_t)(W + (size_t)(n0 + r0) * HID + ce);

  // prologue: stage 0 -> buffer 0
  async_b128(la0, gaA);
  async_b128(la1, gaA + ROWSKIP);
  async_b128(lb0, gaB);
  async_b128(lb1, gaB + ROWSKIP);
  wait_async0();
  __syncthreads();

  v8f acc[2][4];
#pragma unroll
  for (int i = 0; i < 2; ++i)
#pragma unroll
    for (int j = 0; j < 4; ++j)
      acc[i][j] = (v8f){0.f, 0.f, 0.f, 0.f, 0.f, 0.f, 0.f, 0.f};

  const int KIT = HID / TK;  // 128
  for (int ks = 0; ks < KIT; ++ks) {
    const int buf = ks & 1;
    const bool hn = (ks + 1 < KIT);
    if (hn) {  // async-fill the idle buffer while we compute from `buf`
      const unsigned lo = (unsigned)(buf ^ 1) * BUFB;
      const unsigned long long ko = (unsigned long long)(ks + 1) * KBYTES;
      async_b128(la0 + lo, gaA + ko);
      async_b128(la1 + lo, gaA + ko + ROWSKIP);
      async_b128(lb0 + lo, gaB + ko);
      async_b128(lb1 + lo, gaB + ko + ROWSKIP);
    }
    const unsigned char* sA = smem + buf * BUFB;
    const unsigned char* sB = sA + STG;
    v16bf af[2], bg[4];
#pragma unroll
    for (int i = 0; i < 2; ++i) {  // A 16x32: lane-half K chunks 8 apart
      const int row = wm * 32 + i * 16 + (lane & 15);
      af[i] = frag_from(sA + row * RSB + ha * 16,
                        sA + row * RSB + 32 + ha * 16);
    }
#pragma unroll
    for (int j = 0; j < 4; ++j) {  // B 32x16: lane-half K chunks 8 apart
      const int row = wn * 64 + j * 16 + (lane & 15);
      bg[j] = frag_from(sB + row * RSB + ha * 16,
                        sB + row * RSB + 32 + ha * 16);
    }
#pragma unroll
    for (int i = 0; i < 2; ++i)
#pragma unroll
      for (int j = 0; j < 4; ++j)
        acc[i][j] = __builtin_amdgcn_wmma_f32_16x16x32_bf16(
            false, af[i], false, bg[j], (short)0, acc[i][j], false, false);
    if (hn) {
      wait_async0();    // my async fills landed
      __syncthreads();  // everyone's landed; prior readers already drained
    }
  }

  // epilogue: C 16x16 f32 layout -> m = v + (lane>=16)*8, n = lane&15
#pragma unroll
  for (int i = 0; i < 2; ++i)
#pragma unroll
    for (int j = 0; j < 4; ++j) {
      const int n  = n0 + wn * 64 + j * 16 + (lane & 15);
      const int mb = m0 + wm * 32 + i * 16 + ha * 8;
#pragma unroll
      for (int v = 0; v < 8; ++v)
        C[(size_t)(mb + v) * HID + n] = acc[i][j][v];
    }
}

// ---------------------------------------------------------------------------
// fp32 RMSNorm over rows of 4096, emits normalized bf16. One WG per row.
// ---------------------------------------------------------------------------
__global__ __launch_bounds__(256) void xattn_rmsnorm_bf16(
    const float* __restrict__ X, const float* __restrict__ G,
    __bf16* __restrict__ O) {
  const int row = blockIdx.x;
  const int t   = threadIdx.x;
  const float4* xr = (const float4*)(X + (size_t)row * HID);
  float4 v[4];
  float ss = 0.f;
#pragma unroll
  for (int i = 0; i < 4; ++i) {
    v[i] = xr[i * 256 + t];
    ss += v[i].x * v[i].x + v[i].y * v[i].y + v[i].z * v[i].z + v[i].w * v[i].w;
  }
#pragma unroll
  for (int o = 16; o > 0; o >>= 1) ss += __shfl_xor(ss, o, 32);
  __shared__ float red[8];
  if ((t & 31) == 0) red[t >> 5] = ss;
  __syncthreads();
  float tot = 0.f;
#pragma unroll
  for (int w = 0; w < 8; ++w) tot += red[w];
  const float r = rsqrtf(tot * (1.0f / (float)HID) + 1e-6f);
  const float4* gr = (const float4*)G;
  v4bf* orow = (v4bf*)(O + (size_t)row * HID);
#pragma unroll
  for (int i = 0; i < 4; ++i) {
    const float4 g = gr[i * 256 + t];
    v4bf o;
    o[0] = (__bf16)(g.x * (v[i].x * r));
    o[1] = (__bf16)(g.y * (v[i].y * r));
    o[2] = (__bf16)(g.z * (v[i].z * r));
    o[3] = (__bf16)(g.w * (v[i].w * r));
    orow[i * 256 + t] = o;
  }
}

// ---------------------------------------------------------------------------
// out[i,j] = (1/H) * sum_h expf(s - s),  s = <qh[i],kh[j]>/sqrt(128).
// Same WMMA/async pipeline; per-head epilogue every 4 K-steps.
// ---------------------------------------------------------------------------
__global__ __launch_bounds__(256) void xattn_scores(
    const __bf16* __restrict__ Q, const __bf16* __restrict__ K,
    float* __restrict__ OUT) {
  __shared__ unsigned char smem[2 * BUFB];
  const int t    = threadIdx.x;
  const int lane = t & 31;
  const int wid  = t >> 5;
  const int wm   = wid & 3;
  const int wn   = wid >> 2;
  const int m0   = blockIdx.y * TM;
  const int n0   = blockIdx.x * TN;
  const int r0   = t >> 2;
  const int cb   = (t & 3) * 16;
  const int ce   = (t & 3) * 8;
  const int ha   = lane >> 4;
  const float scale = 0.08838834764831845f;  // 1/sqrt(128)

  const unsigned sb  = (unsigned)(uintptr_t)smem;
  const unsigned la0 = sb + r0 * RSB + cb;
  const unsigned la1 = la0 + 64 * RSB;
  const unsigned lb0 = la0 + STG;
  const unsigned lb1 = la1 + STG;
  const unsigned long long gaA =
      (unsigned long long)(uintptr_t)(Q + (size_t)(m0 + r0) * HID + ce);
  const unsigned long long gaB =
      (unsigned long long)(uintptr_t)(K + (size_t)(n0 + r0) * HID + ce);

  async_b128(la0, gaA);
  async_b128(la1, gaA + ROWSKIP);
  async_b128(lb0, gaB);
  async_b128(lb1, gaB + ROWSKIP);
  wait_async0();
  __syncthreads();

  v8f acc[2][4], tot[2][4];
#pragma unroll
  for (int i = 0; i < 2; ++i)
#pragma unroll
    for (int j = 0; j < 4; ++j) {
      acc[i][j] = (v8f){0.f, 0.f, 0.f, 0.f, 0.f, 0.f, 0.f, 0.f};
      tot[i][j] = (v8f){0.f, 0.f, 0.f, 0.f, 0.f, 0.f, 0.f, 0.f};
    }

  const int KIT = HID / TK;  // 128 = 32 heads * 4 steps
  for (int ks = 0; ks < KIT; ++ks) {
    const int buf = ks & 1;
    const bool hn = (ks + 1 < KIT);
    if (hn) {
      const unsigned lo = (unsigned)(buf ^ 1) * BUFB;
      const unsigned long long ko = (unsigned long long)(ks + 1) * KBYTES;
      async_b128(la0 + lo, gaA + ko);
      async_b128(la1 + lo, gaA + ko + ROWSKIP);
      async_b128(lb0 + lo, gaB + ko);
      async_b128(lb1 + lo, gaB + ko + ROWSKIP);
    }
    const unsigned char* sA = smem + buf * BUFB;
    const unsigned char* sB = sA + STG;
    v16bf af[2], bg[4];
#pragma unroll
    for (int i = 0; i < 2; ++i) {
      const int row = wm * 32 + i * 16 + (lane & 15);
      af[i] = frag_from(sA + row * RSB + ha * 16,
                        sA + row * RSB + 32 + ha * 16);
    }
#pragma unroll
    for (int j = 0; j < 4; ++j) {
      const int row = wn * 64 + j * 16 + (lane & 15);
      bg[j] = frag_from(sB + row * RSB + ha * 16,
                        sB + row * RSB + 32 + ha * 16);
    }
#pragma unroll
    for (int i = 0; i < 2; ++i)
#pragma unroll
      for (int j = 0; j < 4; ++j)
        acc[i][j] = __builtin_amdgcn_wmma_f32_16x16x32_bf16(
            false, af[i], false, bg[j], (short)0, acc[i][j], false, false);

    if ((ks & 3) == 3) {  // head boundary: trivial softmax, accumulate, reset
#pragma unroll
      for (int i = 0; i < 2; ++i)
#pragma unroll
        for (int j = 0; j < 4; ++j)
#pragma unroll
          for (int v = 0; v < 8; ++v) {
            const float s = acc[i][j][v] * scale;
            tot[i][j][v] += __expf(s - s);  // == 1 finite, NaN-propagating
            acc[i][j][v] = 0.f;
          }
    }

    if (hn) {
      wait_async0();
      __syncthreads();
    }
  }

  const float invH = 1.0f / (float)HEADS;
#pragma unroll
  for (int i = 0; i < 2; ++i)
#pragma unroll
    for (int j = 0; j < 4; ++j) {
      const int n  = n0 + wn * 64 + j * 16 + (lane & 15);
      const int mb = m0 + wm * 32 + i * 16 + ha * 8;
#pragma unroll
      for (int v = 0; v < 8; ++v)
        OUT[(size_t)(mb + v) * NB2 + n] = tot[i][j][v] * invH;
    }
}

// ---------------------------------------------------------------------------
// Host launch
// ---------------------------------------------------------------------------
extern "C" void kernel_launch(void* const* d_in, const int* in_sizes, int n_in,
                              void* d_out, int out_size, void* d_ws,
                              size_t ws_size, hipStream_t stream) {
  (void)in_sizes; (void)n_in; (void)out_size; (void)ws_size;
  const float* v1 = (const float*)d_in[0];  // (2048, 4096)
  const float* v2 = (const float*)d_in[1];  // (2048, 4096)
  const float* wq = (const float*)d_in[2];  // (4096, 4096)
  const float* wk = (const float*)d_in[3];  // (4096, 4096)
  const float* gq = (const float*)d_in[4];  // (4096,)
  const float* gk = (const float*)d_in[5];  // (4096,)
  float* out = (float*)d_out;               // (2048, 2048)

  // Workspace layout (160 MB; qn/kn reuse weight scratch, dead after GEMMs):
  const size_t MB = 1ull << 20;
  char* ws = (char*)d_ws;
  __bf16* v1b = (__bf16*)(ws + 0 * MB);    // 16 MB
  __bf16* v2b = (__bf16*)(ws + 16 * MB);   // 16 MB
  __bf16* wqb = (__bf16*)(ws + 32 * MB);   // 32 MB
  __bf16* wkb = (__bf16*)(ws + 64 * MB);   // 32 MB
  float*  qf  = (float*)(ws + 96 * MB);    // 32 MB
  float*  kf  = (float*)(ws + 128 * MB);   // 32 MB
  __bf16* qnb = (__bf16*)(ws + 32 * MB);   // aliases wqb
  __bf16* knb = (__bf16*)(ws + 64 * MB);   // aliases wkb

  const int nv4 = (NB1 * HID) / 4;
  const int nw4 = (HID * HID) / 4;

  xattn_f32_to_bf16<<<(nv4 + 255) / 256, 256, 0, stream>>>(v1, v1b, nv4);
  xattn_f32_to_bf16<<<(nv4 + 255) / 256, 256, 0, stream>>>(v2, v2b, nv4);
  xattn_f32_to_bf16<<<(nw4 + 255) / 256, 256, 0, stream>>>(wq, wqb, nw4);
  xattn_f32_to_bf16<<<(nw4 + 255) / 256, 256, 0, stream>>>(wk, wkb, nw4);

  dim3 gproj(HID / TN, NB1 / TM);  // (32, 16)
  xattn_gemm_proj<<<gproj, 256, 0, stream>>>(v1b, wqb, qf);
  xattn_gemm_proj<<<gproj, 256, 0, stream>>>(v2b, wkb, kf);

  xattn_rmsnorm_bf16<<<NB1, 256, 0, stream>>>(qf, gq, qnb);
  xattn_rmsnorm_bf16<<<NB2, 256, 0, stream>>>(kf, gk, knb);

  dim3 gsc(NB2 / TN, NB1 / TM);    // (16, 16)
  xattn_scores<<<gsc, 256, 0, stream>>>(qnb, knb, out);
}
